// MoELayer_18519898980911
// MI455X (gfx1250) — matrix operand order repaired and checked by reading, not compiled, not run on hardware
//
#include <hip/hip_runtime.h>
#include <hip/hip_bf16.h>
#include <cstddef>
#include <cstdint>

// ---------------- problem constants (match reference) ----------------
#define D_MODEL   1024
#define D_FF      4096
#define N_EXPERTS 8
#define TOKENS    (8 * 2048)                      // B*N
#define SLOT_CAP  (2 * TOKENS + 128 * N_EXPERTS)  // top-2 slots + per-expert pad to 128
#define M_TILES   (SLOT_CAP / 128)                // 264

// ctrl int layout in workspace:
//   [0..7]  counts      [8..15] bases (128-aligned)
//   [16..23] cursors    [24]    total padded rows (multiple of 128)
#define CTRL_COUNT  0
#define CTRL_BASE   8
#define CTRL_CUR    16
#define CTRL_TOTAL  24

// LDS B-panel: 128 rows x 32 k (bf16), 80B row pitch (64B data + 16B pad for banks)
#define BPITCH 80

typedef __bf16        v16bf __attribute__((ext_vector_type(16)));
typedef float         v8f   __attribute__((ext_vector_type(8)));
typedef unsigned int  u32x4 __attribute__((ext_vector_type(4)));

union Frag { v16bf v; u32x4 q[2]; };

__device__ __forceinline__ float gelu_exact(float v) {
    return 0.5f * v * (1.0f + erff(v * 0.70710678118654752f));
}

// Cooperative stage of a 128-row x 32-k bf16 B panel into LDS.
// 512 16B chunks, 256 threads x 2 chunks. chunk id -> row = id/4, c = id%4.
__device__ __forceinline__ void stage_b(const __bf16* __restrict__ bsrc, size_t row_stride,
                                        int k0, unsigned char* __restrict__ lds, int tid) {
#pragma unroll
    for (int i = 0; i < 2; ++i) {
        int id = tid + i * 256;
        int r = id >> 2, c = id & 3;
        u32x4 v = *(const u32x4*)(bsrc + (size_t)r * row_stride + k0 + c * 8);
        *(u32x4*)(lds + r * BPITCH + c * 16) = v;
    }
}

// ---------------- utility kernels ----------------
__global__ void zero_f32_kernel(float* __restrict__ p, size_t n) {
    size_t i = (size_t)blockIdx.x * blockDim.x + threadIdx.x;
    size_t stride = (size_t)gridDim.x * blockDim.x;
    for (; i < n; i += stride) p[i] = 0.0f;
}

__global__ void zero_i32_kernel(int* __restrict__ p, int n) {
    int i = blockIdx.x * blockDim.x + threadIdx.x;
    if (i < n) p[i] = 0;
}

__global__ void cast_bf16_kernel(const float* __restrict__ s, __bf16* __restrict__ d, size_t n) {
    size_t i = (size_t)blockIdx.x * blockDim.x + threadIdx.x;
    size_t stride = (size_t)gridDim.x * blockDim.x;
    for (; i < n; i += stride) d[i] = (__bf16)s[i];
}

// ---------------- routing ----------------
// One wave32 per token: coalesced dot products over C for all 8 experts.
__global__ __launch_bounds__(256)
void router_kernel(const float* __restrict__ x, const float* __restrict__ rw,
                   int* __restrict__ topk_idx, float* __restrict__ topk_gate,
                   int* __restrict__ ctrl) {
    int t    = blockIdx.x * 8 + (threadIdx.x >> 5);
    int lane = threadIdx.x & 31;
    if (t >= TOKENS) return;

    const float* xr = x + (size_t)t * D_MODEL;
    float acc[N_EXPERTS];
#pragma unroll
    for (int e = 0; e < N_EXPERTS; ++e) acc[e] = 0.0f;

    for (int c = lane; c < D_MODEL; c += 32) {
        float xv = xr[c];
#pragma unroll
        for (int e = 0; e < N_EXPERTS; ++e)
            acc[e] += xv * rw[e * D_MODEL + c];
    }
#pragma unroll
    for (int e = 0; e < N_EXPERTS; ++e)
        for (int off = 16; off > 0; off >>= 1)
            acc[e] += __shfl_xor(acc[e], off, 32);

    if (lane == 0) {
        float m = acc[0];
#pragma unroll
        for (int e = 1; e < N_EXPERTS; ++e) m = fmaxf(m, acc[e]);
        float p[N_EXPERTS], sum = 0.0f;
#pragma unroll
        for (int e = 0; e < N_EXPERTS; ++e) { p[e] = expf(acc[e] - m); sum += p[e]; }
        float inv = 1.0f / sum;
#pragma unroll
        for (int e = 0; e < N_EXPERTS; ++e) p[e] *= inv;

        int i0 = 0;
#pragma unroll
        for (int e = 1; e < N_EXPERTS; ++e) if (p[e] > p[i0]) i0 = e;
        int i1 = (i0 == 0) ? 1 : 0;
#pragma unroll
        for (int e = 0; e < N_EXPERTS; ++e)
            if (e != i0 && p[e] > p[i1]) i1 = e;

        float wn = 1.0f / (p[i0] + p[i1] + 1e-9f);
        topk_idx[2 * t]      = i0;
        topk_idx[2 * t + 1]  = i1;
        topk_gate[2 * t]     = p[i0] * wn;
        topk_gate[2 * t + 1] = p[i1] * wn;
        atomicAdd(&ctrl[CTRL_COUNT + i0], 1);
        atomicAdd(&ctrl[CTRL_COUNT + i1], 1);
    }
}

__global__ void scan_kernel(int* __restrict__ ctrl) {
    if (blockIdx.x == 0 && threadIdx.x == 0) {
        int tot = 0;
        for (int e = 0; e < N_EXPERTS; ++e) {
            ctrl[CTRL_BASE + e] = tot;
            tot += (ctrl[CTRL_COUNT + e] + 127) & ~127;   // 128-align expert regions
        }
        ctrl[CTRL_TOTAL] = tot;
    }
}

__global__ void fill_kernel(const int* __restrict__ topk_idx, const float* __restrict__ topk_gate,
                            int* __restrict__ ctrl, int* __restrict__ slot_tok,
                            int* __restrict__ slot_exp, float* __restrict__ slot_gate) {
    int t = blockIdx.x * blockDim.x + threadIdx.x;
    if (t >= TOKENS) return;
#pragma unroll
    for (int k = 0; k < 2; ++k) {
        int e   = topk_idx[2 * t + k];
        int pos = atomicAdd(&ctrl[CTRL_CUR + e], 1);
        int s   = ctrl[CTRL_BASE + e] + pos;
        slot_tok[s]  = t;
        slot_exp[s]  = e;
        slot_gate[s] = topk_gate[2 * t + k];
    }
}

__global__ void pad_kernel(const int* __restrict__ ctrl, int* __restrict__ slot_tok,
                           int* __restrict__ slot_exp, float* __restrict__ slot_gate) {
    int e = blockIdx.x;
    int c = ctrl[CTRL_COUNT + e];
    int pc = (c + 127) & ~127;
    int i = threadIdx.x;
    if (c + i < pc) {
        int s = ctrl[CTRL_BASE + e] + c + i;
        slot_tok[s]  = 0;
        slot_exp[s]  = e;
        slot_gate[s] = 0.0f;   // padded rows contribute nothing
    }
}

// ---------------- GEMM1: h = gelu(x_gathered @ w1^T + b1) ----------------
// Workgroup (8 waves) computes a 128x128 tile; each wave 16 rows x 128 cols.
// B panel (128 f-rows x 32 k) double-buffered in LDS, shared by all 8 waves.
__global__ __launch_bounds__(256)
void gemm1_kernel(const __bf16* __restrict__ xbf, const __bf16* __restrict__ w1,
                  const float* __restrict__ b1, const int* __restrict__ slot_tok,
                  const int* __restrict__ slot_exp, const int* __restrict__ ctrl,
                  __bf16* __restrict__ h) {
    __shared__ __align__(16) unsigned char smem[2][128 * BPITCH];
    const int NTW = D_FF / 128;          // 32 column tiles
    int mt = blockIdx.x / NTW, nt = blockIdx.x % NTW;
    int srow0 = mt * 128;
    if (srow0 >= ctrl[CTRL_TOTAL]) return;   // workgroup-uniform, before any barrier

    int tid  = threadIdx.x;
    int wave = tid >> 5, lane = tid & 31;
    int half = lane >> 4, lid = lane & 15;

    int e    = slot_exp[srow0];              // uniform: expert regions 128-aligned
    int mrow = srow0 + wave * 16;
    int tok  = slot_tok[mrow + lid];         // A-fragment row for this lane
    const __bf16* arow = xbf + (size_t)tok * D_MODEL;
    const __bf16* bsrc = w1 + ((size_t)e * D_FF + (size_t)nt * 128) * D_MODEL;

    v8f acc[8] = {};
    int buf = 0;
    stage_b(bsrc, D_MODEL, 0, smem[0], tid);
    __syncthreads();
    for (int k0 = 0; k0 < D_MODEL; k0 += 32) {
        if (k0 + 32 < D_MODEL) stage_b(bsrc, D_MODEL, k0 + 32, smem[buf ^ 1], tid);
        Frag a;
        const __bf16* ap = arow + k0 + half * 16;  // A: lanes<16 K{0..15}; lanes>=16 K{16..31}
        a.q[0] = *(const u32x4*)(ap);
        a.q[1] = *(const u32x4*)(ap + 8);
#pragma unroll
        for (int s = 0; s < 8; ++s) {
            Frag b;
            const unsigned char* bb = smem[buf] + (s * 16 + lid) * BPITCH + half * 32;
            b.q[0] = *(const u32x4*)(bb);
            b.q[1] = *(const u32x4*)(bb + 16);
            acc[s] = __builtin_amdgcn_wmma_f32_16x16x32_bf16(
                false, a.v, false, b.v, (short)0, acc[s], false, false);
        }
        __syncthreads();
        buf ^= 1;
    }

#pragma unroll
    for (int s = 0; s < 8; ++s) {
        int col = nt * 128 + s * 16 + lid;         // D: N = lane&15
        float bias = b1[(size_t)e * D_FF + col];
#pragma unroll
        for (int r = 0; r < 8; ++r) {
            float v = gelu_exact(acc[s][r] + bias);
            h[(size_t)(mrow + r + half * 8) * D_FF + col] = (__bf16)v;
        }
    }
}

// ---------------- GEMM2: out[tok] += gate * (h @ w2^T + b2) ----------------
__global__ __launch_bounds__(256)
void gemm2_kernel(const __bf16* __restrict__ h, const __bf16* __restrict__ w2,
                  const float* __restrict__ b2, const int* __restrict__ slot_tok,
                  const int* __restrict__ slot_exp, const float* __restrict__ slot_gate,
                  const int* __restrict__ ctrl, float* __restrict__ out) {
    __shared__ __align__(16) unsigned char smem[2][128 * BPITCH];
    const int NTW = D_MODEL / 128;       // 8 column tiles
    int mt = blockIdx.x / NTW, nt = blockIdx.x % NTW;
    int srow0 = mt * 128;
    if (srow0 >= ctrl[CTRL_TOTAL]) return;

    int tid  = threadIdx.x;
    int wave = tid >> 5, lane = tid & 31;
    int half = lane >> 4, lid = lane & 15;

    int e    = slot_exp[srow0];
    int mrow = srow0 + wave * 16;
    const __bf16* arow = h + (size_t)(mrow + lid) * D_FF;
    const __bf16* bsrc = w2 + ((size_t)e * D_MODEL + (size_t)nt * 128) * D_FF;

    v8f acc[8] = {};
    int buf = 0;
    stage_b(bsrc, D_FF, 0, smem[0], tid);
    __syncthreads();
    for (int k0 = 0; k0 < D_FF; k0 += 32) {
        if (k0 + 32 < D_FF) stage_b(bsrc, D_FF, k0 + 32, smem[buf ^ 1], tid);
        Frag a;
        const __bf16* ap = arow + k0 + half * 16;
        a.q[0] = *(const u32x4*)(ap);
        a.q[1] = *(const u32x4*)(ap + 8);
#pragma unroll
        for (int s = 0; s < 8; ++s) {
            Frag b;
            const unsigned char* bb = smem[buf] + (s * 16 + lid) * BPITCH + half * 32;
            b.q[0] = *(const u32x4*)(bb);
            b.q[1] = *(const u32x4*)(bb + 16);
            acc[s] = __builtin_amdgcn_wmma_f32_16x16x32_bf16(
                false, a.v, false, b.v, (short)0, acc[s], false, false);
        }
        __syncthreads();
        buf ^= 1;
    }

    int   tokM[8];
    float gM[8];
#pragma unroll
    for (int r = 0; r < 8; ++r) {
        int s2  = mrow + r + half * 8;
        tokM[r] = slot_tok[s2];
        gM[r]   = slot_gate[s2];
    }
#pragma unroll
    for (int s = 0; s < 8; ++s) {
        int col = nt * 128 + s * 16 + lid;
        float b2v = b2[(size_t)e * D_MODEL + col];
#pragma unroll
        for (int r = 0; r < 8; ++r) {
            float v = gM[r] * (acc[s][r] + b2v);
            // exactly two contributions per out element -> bitwise deterministic
            unsafeAtomicAdd(&out[(size_t)tokM[r] * D_MODEL + col], v);
        }
    }
}

// ---------------- host launcher ----------------
extern "C" void kernel_launch(void* const* d_in, const int* in_sizes, int n_in,
                              void* d_out, int out_size, void* d_ws, size_t ws_size,
                              hipStream_t stream) {
    (void)in_sizes; (void)n_in; (void)out_size; (void)ws_size;

    const float* x    = (const float*)d_in[0];
    const float* rw   = (const float*)d_in[1];
    const float* w1f  = (const float*)d_in[2];
    const float* b1   = (const float*)d_in[3];
    const float* w2f  = (const float*)d_in[4];
    const float* b2   = (const float*)d_in[5];
    float*       out  = (float*)d_out;

    // workspace carve-up (256B aligned)
    uint8_t* ws = (uint8_t*)d_ws;
    size_t off = 0;
    auto alloc = [&](size_t bytes) -> uint8_t* {
        uint8_t* p = ws + off;
        off = (off + bytes + 255) & ~(size_t)255;
        return p;
    };
    __bf16* xbf   = (__bf16*)alloc((size_t)TOKENS * D_MODEL * 2);
    __bf16* w1bf  = (__bf16*)alloc((size_t)N_EXPERTS * D_FF * D_MODEL * 2);
    __bf16* w2bf  = (__bf16*)alloc((size_t)N_EXPERTS * D_MODEL * D_FF * 2);
    __bf16* hbuf  = (__bf16*)alloc((size_t)SLOT_CAP * D_FF * 2);
    int*    tkidx = (int*)alloc((size_t)TOKENS * 2 * 4);
    float*  tkg   = (float*)alloc((size_t)TOKENS * 2 * 4);
    int*    stok  = (int*)alloc((size_t)SLOT_CAP * 4);
    int*    sexp  = (int*)alloc((size_t)SLOT_CAP * 4);
    float*  sgate = (float*)alloc((size_t)SLOT_CAP * 4);
    int*    ctrl  = (int*)alloc(32 * 4);

    // 1. zero output (atomic accumulation target) + control ints
    zero_f32_kernel<<<4096, 256, 0, stream>>>(out, (size_t)TOKENS * D_MODEL);
    zero_i32_kernel<<<1, 32, 0, stream>>>(ctrl, 32);

    // 2. fp32 -> bf16 casts (bf16 weight set ~134MB -> resident in 192MB L2)
    cast_bf16_kernel<<<4096, 256, 0, stream>>>(x,   xbf,  (size_t)TOKENS * D_MODEL);
    cast_bf16_kernel<<<8192, 256, 0, stream>>>(w1f, w1bf, (size_t)N_EXPERTS * D_FF * D_MODEL);
    cast_bf16_kernel<<<8192, 256, 0, stream>>>(w2f, w2bf, (size_t)N_EXPERTS * D_MODEL * D_FF);

    // 3. routing: softmax + top-2 + counts; compacted per-expert slot lists
    router_kernel<<<TOKENS / 8, 256, 0, stream>>>(x, rw, tkidx, tkg, ctrl);
    scan_kernel<<<1, 1, 0, stream>>>(ctrl);
    fill_kernel<<<(TOKENS + 255) / 256, 256, 0, stream>>>(tkidx, tkg, ctrl, stok, sexp, sgate);
    pad_kernel<<<N_EXPERTS, 128, 0, stream>>>(ctrl, stok, sexp, sgate);

    // 4. expert GEMMs (fixed grids cover worst case; workgroups past total early-exit)
    gemm1_kernel<<<M_TILES * (D_FF / 128),    256, 0, stream>>>(xbf, w1bf, b1, stok, sexp, ctrl, hbuf);
    gemm2_kernel<<<M_TILES * (D_MODEL / 128), 256, 0, stream>>>(hbuf, w2bf, b2, stok, sexp, sgate, ctrl, out);
}
